// SimplifiedLinearAttention_42485816492251
// MI455X (gfx1250) — compile-verified, matrix operand order checked by reading, not run on hardware
//
#include <hip/hip_runtime.h>

// ---------------------------------------------------------------------------
// Types for CDNA5 WMMA (wave32, 16x16x32 bf16 -> f32 accumulate)
// ---------------------------------------------------------------------------
typedef __bf16 bhalf;
typedef __attribute__((ext_vector_type(16))) __bf16 v16bf;
typedef __attribute__((ext_vector_type(8)))  __bf16 v8bf;
typedef __attribute__((ext_vector_type(8)))  float  v8f;

// Problem constants
#define BATCH    8
#define NTOK     3136         // 56*56
#define CDIM     512
#define NH       8
#define HD       64           // head dim
#define BH       (BATCH*NH)   // 64
#define MROWS    (BATCH*NTOK) // 25088
#define QKVN     (3*CDIM)     // 1536

static __device__ __forceinline__ float bf2f(bhalf h) { return (float)h; }
static __device__ __forceinline__ bhalf f2bf(float f) { return (bhalf)f; }

static __device__ __forceinline__ v8f
wmma_bf16(v16bf a, v16bf b, v8f c) {
    return __builtin_amdgcn_wmma_f32_16x16x32_bf16(false, a, false, b, (short)0, c,
                                                   false, false);
}

// Load a 16-lane-row, K-major WMMA fragment (A or B) from a row-major
// [rows][ld] bf16 matrix. Lane l in [0,16): holds K = k0..k0+7 (elems 0..7)
// and K = k0+16..k0+23 (elems 8..15). Lane l in [16,32): K offset +8.
// Requires ld % 8 == 0 and 16B-aligned base.
static __device__ __forceinline__ v16bf
load_frag_kmajor(const bhalf* __restrict__ base, int row, int ld, int k0, int lane) {
    const bhalf* p = base + (size_t)row * (size_t)ld + (size_t)(k0 + ((lane >> 4) << 3));
    v8bf lo = *(const v8bf*)(p);
    v8bf hi = *(const v8bf*)(p + 16);
    v16bf r;
#pragma unroll
    for (int i = 0; i < 8; ++i) { r[i] = lo[i]; r[i + 8] = hi[i]; }
    return r;
}

// ---------------------------------------------------------------------------
// K0a: float -> bf16 straight convert (x)
// ---------------------------------------------------------------------------
__global__ void cvt_f2b_kernel(const float* __restrict__ src, bhalf* __restrict__ dst, int nElem) {
    int i = blockIdx.x * blockDim.x + threadIdx.x;
    if (i < nElem) dst[i] = f2bf(src[i]);
}

// K0b: transpose+convert: src[rows][cols] f32 -> dst[cols][rows] bf16
__global__ void transpose_cvt_kernel(const float* __restrict__ src, bhalf* __restrict__ dst,
                                     int rows, int cols) {
    int i = blockIdx.x * blockDim.x + threadIdx.x;
    if (i < rows * cols) {
        int r = i / cols;
        int c = i - r * cols;
        dst[(size_t)c * rows + r] = f2bf(src[i]);
    }
}

// ---------------------------------------------------------------------------
// K1: fused QKV GEMM  Y = x @ [Wq | Wk | Wv]  (M=25088, N=1536, K=512)
// Register-blocked: each wave computes a 32(M) x 64(N) macro-tile
// (MT=2 x NT=4 WMMA tiles) -> A frags reused 4x, B frags reused 2x.
//   q = relu(Y[:, :512])          -> q  [bh][n][d]
//   k = relu(Y[:, 512:1024]+pos)  -> kT [bh][d][n]
//   v = Y[:, 1024:]               -> vT [bh][d][n]   (NCHW fmap layout)
// ---------------------------------------------------------------------------
__global__ __launch_bounds__(256)
void qkv_gemm_kernel(const bhalf* __restrict__ xb,     // [25088][512]
                     const bhalf* __restrict__ wqkvT,  // [1536][512]
                     const float* __restrict__ pos,    // [3136][512]
                     bhalf* __restrict__ q,
                     bhalf* __restrict__ kT,
                     bhalf* __restrict__ vT) {
    const int lane = threadIdx.x & 31;
    const int wave = threadIdx.x >> 5;
    const int n0 = (blockIdx.x * 8 + wave) * 64;  // wave macro-col (24 total)
    const int m0 = blockIdx.y * 32;               // wave macro-row (784 total)

    v8f acc[2][4] = {};
#pragma unroll 2
    for (int k0 = 0; k0 < CDIM; k0 += 32) {
        v16bf a[2], b[4];
#pragma unroll
        for (int mt = 0; mt < 2; ++mt)
            a[mt] = load_frag_kmajor(xb, m0 + mt * 16 + (lane & 15), CDIM, k0, lane);
#pragma unroll
        for (int nt = 0; nt < 4; ++nt)
            b[nt] = load_frag_kmajor(wqkvT, n0 + nt * 16 + (lane & 15), CDIM, k0, lane);
#pragma unroll
        for (int mt = 0; mt < 2; ++mt)
#pragma unroll
            for (int nt = 0; nt < 4; ++nt)
                acc[mt][nt] = wmma_bf16(a[mt], b[nt], acc[mt][nt]);
    }

#pragma unroll
    for (int mt = 0; mt < 2; ++mt) {
#pragma unroll
        for (int nt = 0; nt < 4; ++nt) {
            const int col = n0 + nt * 16 + (lane & 15);  // output channel [0,1536)
#pragma unroll
            for (int r = 0; r < 8; ++r) {
                int m    = m0 + mt * 16 + r + ((lane >> 4) << 3);
                int bidx = m / NTOK;
                int n    = m - bidx * NTOK;
                float val = acc[mt][nt][r];
                if (col < CDIM) {                       // Q, relu
                    val = fmaxf(val, 0.0f);
                    int h = col >> 6, d = col & 63;
                    q[(((size_t)(bidx * NH + h)) * NTOK + n) * HD + d] = f2bf(val);
                } else if (col < 2 * CDIM) {            // K, +pos, relu, transposed
                    int cc = col - CDIM;
                    val += pos[(size_t)n * CDIM + cc];
                    val = fmaxf(val, 0.0f);
                    int h = cc >> 6, d = cc & 63;
                    kT[(((size_t)(bidx * NH + h)) * HD + d) * NTOK + n] = f2bf(val);
                } else {                                // V, transposed
                    int cc = col - 2 * CDIM;
                    int h = cc >> 6, d = cc & 63;
                    vT[(((size_t)(bidx * NH + h)) * HD + d) * NTOK + n] = f2bf(val);
                }
            }
        }
    }
}

// ---------------------------------------------------------------------------
// K2: ksum[bh][c] = sum_n k[bh][n][c]   (rows of kT are contiguous)
// ---------------------------------------------------------------------------
__global__ __launch_bounds__(256)
void ksum_kernel(const bhalf* __restrict__ kT, float* __restrict__ ksum) {
    __shared__ float red[256];
    const int bh = blockIdx.x;
    const int t = threadIdx.x;
    const int c = t >> 2;       // channel 0..63
    const int part = t & 3;     // quarter of N
    const bhalf* row = kT + ((size_t)bh * HD + c) * NTOK;
    float s = 0.0f;
    const int n0 = part * (NTOK / 4);
    for (int n = n0; n < n0 + NTOK / 4; ++n) s += bf2f(row[n]);
    red[t] = s;
    __syncthreads();
    if (part == 0) ksum[bh * HD + c] = red[t] + red[t + 1] + red[t + 2] + red[t + 3];
}

// ---------------------------------------------------------------------------
// K3: kvmat[bh][c][d] = sum_n k[n][c] * v[n][d]   (K = 3136 reduction)
// Register-blocked 2x2: each wave computes 32(c) x 32(d); 4 waves per head.
// D stored transposed -> kvT[bh][d][c] (K-major B frag for the out-GEMM).
// ---------------------------------------------------------------------------
__global__ __launch_bounds__(256)
void kvmat_kernel(const bhalf* __restrict__ kT, const bhalf* __restrict__ vT,
                  bhalf* __restrict__ kvT) {
    const int lane = threadIdx.x & 31;
    const int wave = threadIdx.x >> 5;
    const int id = blockIdx.x * 8 + wave;     // 0..255
    const int bh = id >> 2;
    const int tile = id & 3;
    const int m0 = (tile >> 1) * 32;          // c macro-tile
    const int n0 = (tile & 1) * 32;           // d macro-tile
    const bhalf* A = kT + (size_t)bh * HD * NTOK;
    const bhalf* B = vT + (size_t)bh * HD * NTOK;

    v8f acc[2][2] = {};
    for (int k0 = 0; k0 < NTOK; k0 += 32) {
        if (k0 + 32 < NTOK) {  // warm caches for the next K step
            __builtin_prefetch(A + (size_t)(m0 + (lane & 15)) * NTOK + k0 + 32, 0, 0);
            __builtin_prefetch(B + (size_t)(n0 + (lane & 15)) * NTOK + k0 + 32, 0, 0);
        }
        v16bf a[2], b[2];
#pragma unroll
        for (int mt = 0; mt < 2; ++mt)
            a[mt] = load_frag_kmajor(A, m0 + mt * 16 + (lane & 15), NTOK, k0, lane);
#pragma unroll
        for (int nt = 0; nt < 2; ++nt)
            b[nt] = load_frag_kmajor(B, n0 + nt * 16 + (lane & 15), NTOK, k0, lane);
#pragma unroll
        for (int mt = 0; mt < 2; ++mt)
#pragma unroll
            for (int nt = 0; nt < 2; ++nt)
                acc[mt][nt] = wmma_bf16(a[mt], b[nt], acc[mt][nt]);
    }
#pragma unroll
    for (int mt = 0; mt < 2; ++mt) {
#pragma unroll
        for (int nt = 0; nt < 2; ++nt) {
            const int d = n0 + nt * 16 + (lane & 15);
#pragma unroll
            for (int r = 0; r < 8; ++r) {
                int c = m0 + mt * 16 + r + ((lane >> 4) << 3);
                kvT[(size_t)bh * (HD * HD) + d * HD + c] = f2bf(acc[mt][nt][r]);
            }
        }
    }
}

// ---------------------------------------------------------------------------
// K4: z[bh][n] = 1 / (dot(q[bh][n][:], ksum[bh][:]) + 1e-6)
// ---------------------------------------------------------------------------
__global__ __launch_bounds__(256)
void z_kernel(const bhalf* __restrict__ q, const float* __restrict__ ksum,
              float* __restrict__ z) {
    __shared__ float ks[HD];
    const int bh = blockIdx.x;
    if (threadIdx.x < HD) ks[threadIdx.x] = ksum[bh * HD + threadIdx.x];
    __syncthreads();
    for (int n = threadIdx.x; n < NTOK; n += 256) {
        const bhalf* row = q + ((size_t)bh * NTOK + n) * HD;
        float s = 0.0f;
#pragma unroll
        for (int c = 0; c < HD; ++c) s += bf2f(row[c]) * ks[c];
        z[bh * NTOK + n] = 1.0f / (s + 1e-6f);
    }
}

// ---------------------------------------------------------------------------
// K5: depthwise 5x5 conv (SAME) over vT viewed as [bh][d][56][56].
// One block per (bh,d) channel; channel staged in LDS. Writes conv result
// (plus bias) into the merged attn buffer; K6 accumulates on top of it.
// ---------------------------------------------------------------------------
__global__ __launch_bounds__(256)
void dwconv_kernel(const bhalf* __restrict__ vT, const float* __restrict__ w,
                   const float* __restrict__ bias, bhalf* __restrict__ attn) {
    __shared__ float fm[NTOK];
    const int bh = blockIdx.x >> 6;
    const int d  = blockIdx.x & 63;
    const bhalf* ch = vT + ((size_t)bh * HD + d) * NTOK;
    for (int i = threadIdx.x; i < NTOK; i += 256) fm[i] = bf2f(ch[i]);
    float wv[25];
#pragma unroll
    for (int i = 0; i < 25; ++i) wv[i] = w[d * 25 + i];
    const float bz = bias[d];
    __syncthreads();

    const int b_ = bh >> 3;
    const int h  = bh & 7;
    for (int idx = threadIdx.x; idx < NTOK; idx += 256) {
        const int y = idx / 56;
        const int x = idx - y * 56;
        float acc = bz;
#pragma unroll
        for (int ky = 0; ky < 5; ++ky) {
            int yy = y + ky - 2;
            if (yy < 0 || yy >= 56) continue;
#pragma unroll
            for (int kx = 0; kx < 5; ++kx) {
                int xx = x + kx - 2;
                if (xx < 0 || xx >= 56) continue;
                acc += fm[yy * 56 + xx] * wv[ky * 5 + kx];
            }
        }
        attn[((size_t)b_ * NTOK + idx) * CDIM + h * HD + d] = f2bf(acc);
    }
}

// ---------------------------------------------------------------------------
// K6: out[bh][n][d] = z[n] * (q[bh] @ kvmat[bh])[n][d] + conv (already in attn)
// Register-blocked 2x2: each wave computes 32(n) x 32(d); K=64 fully unrolled.
// ---------------------------------------------------------------------------
__global__ __launch_bounds__(256)
void attnout_kernel(const bhalf* __restrict__ q, const bhalf* __restrict__ kvT,
                    const float* __restrict__ z, bhalf* __restrict__ attn) {
    const int lane = threadIdx.x & 31;
    const int wave = threadIdx.x >> 5;
    const int id = blockIdx.x * 8 + wave;        // 0..12543
    const int bh = id / 196;
    const int rem = id - bh * 196;
    const int m0 = (rem >> 1) * 32;              // n macro-tile (98 per head)
    const int n0 = (rem & 1) * 32;               // d macro-tile (2 per head)
    const bhalf* A = q   + (size_t)bh * NTOK * HD;
    const bhalf* B = kvT + (size_t)bh * (HD * HD);

    v8f acc[2][2] = {};
#pragma unroll
    for (int k0 = 0; k0 < HD; k0 += 32) {
        v16bf a[2], b[2];
#pragma unroll
        for (int mt = 0; mt < 2; ++mt)
            a[mt] = load_frag_kmajor(A, m0 + mt * 16 + (lane & 15), HD, k0, lane);
#pragma unroll
        for (int nt = 0; nt < 2; ++nt)
            b[nt] = load_frag_kmajor(B, n0 + nt * 16 + (lane & 15), HD, k0, lane);
#pragma unroll
        for (int mt = 0; mt < 2; ++mt)
#pragma unroll
            for (int nt = 0; nt < 2; ++nt)
                acc[mt][nt] = wmma_bf16(a[mt], b[nt], acc[mt][nt]);
    }
    const int b_ = bh >> 3;
    const int h  = bh & 7;
#pragma unroll
    for (int mt = 0; mt < 2; ++mt) {
#pragma unroll
        for (int nt = 0; nt < 2; ++nt) {
            const int d = n0 + nt * 16 + (lane & 15);
#pragma unroll
            for (int r = 0; r < 8; ++r) {
                int n = m0 + mt * 16 + r + ((lane >> 4) << 3);
                float val = acc[mt][nt][r] * z[bh * NTOK + n];
                size_t addr = ((size_t)b_ * NTOK + n) * CDIM + h * HD + d;
                val += bf2f(attn[addr]);   // conv path written by K5
                attn[addr] = f2bf(val);
            }
        }
    }
}

// ---------------------------------------------------------------------------
// K7: final projection  out = attn @ Wproj + bproj   (M=25088, N=512, K=512)
// Register-blocked 2x4: wave = 32(M) x 64(N); one block's 8 waves span N=512.
// ---------------------------------------------------------------------------
__global__ __launch_bounds__(256)
void proj_gemm_kernel(const bhalf* __restrict__ attn, const bhalf* __restrict__ wT,
                      const float* __restrict__ bias, float* __restrict__ out) {
    const int lane = threadIdx.x & 31;
    const int wave = threadIdx.x >> 5;
    const int n0 = wave * 64;              // 8 waves * 64 = 512 output cols
    const int m0 = blockIdx.x * 32;        // 784 macro-rows

    v8f acc[2][4] = {};
#pragma unroll 2
    for (int k0 = 0; k0 < CDIM; k0 += 32) {
        v16bf a[2], b[4];
#pragma unroll
        for (int mt = 0; mt < 2; ++mt)
            a[mt] = load_frag_kmajor(attn, m0 + mt * 16 + (lane & 15), CDIM, k0, lane);
#pragma unroll
        for (int nt = 0; nt < 4; ++nt)
            b[nt] = load_frag_kmajor(wT, n0 + nt * 16 + (lane & 15), CDIM, k0, lane);
#pragma unroll
        for (int mt = 0; mt < 2; ++mt)
#pragma unroll
            for (int nt = 0; nt < 4; ++nt)
                acc[mt][nt] = wmma_bf16(a[mt], b[nt], acc[mt][nt]);
    }
#pragma unroll
    for (int mt = 0; mt < 2; ++mt) {
#pragma unroll
        for (int nt = 0; nt < 4; ++nt) {
            const int col = n0 + nt * 16 + (lane & 15);
            const float bv = bias[col];
#pragma unroll
            for (int r = 0; r < 8; ++r) {
                int m = m0 + mt * 16 + r + ((lane >> 4) << 3);
                out[(size_t)m * CDIM + col] = acc[mt][nt][r] + bv;
            }
        }
    }
}

// ---------------------------------------------------------------------------
// Host launcher
// ---------------------------------------------------------------------------
extern "C" void kernel_launch(void* const* d_in, const int* in_sizes, int n_in,
                              void* d_out, int out_size, void* d_ws, size_t ws_size,
                              hipStream_t stream) {
    (void)in_sizes; (void)n_in; (void)out_size; (void)ws_size;

    const float* x       = (const float*)d_in[0];
    // d_in[1] = H (56), d_in[2] = W (56) — compile-time constants here
    const float* Wq      = (const float*)d_in[3];
    const float* Wkv     = (const float*)d_in[4];
    const float* pos_enc = (const float*)d_in[5];
    const float* dwc_w   = (const float*)d_in[6];
    const float* dwc_b   = (const float*)d_in[7];
    const float* Wproj   = (const float*)d_in[8];
    const float* bproj   = (const float*)d_in[9];
    float* out = (float*)d_out;

    // Workspace carve-out (256B aligned)
    char* ws = (char*)d_ws;
    size_t off = 0;
    auto carve = [&](size_t bytes) -> void* {
        void* p = ws + off;
        off = (off + bytes + 255) & ~(size_t)255;
        return p;
    };
    bhalf* xb     = (bhalf*)carve((size_t)MROWS * CDIM * 2);       // x in bf16
    bhalf* wqkvT  = (bhalf*)carve((size_t)QKVN * CDIM * 2);        // [1536][512]
    bhalf* wprojT = (bhalf*)carve((size_t)CDIM * CDIM * 2);        // [512][512]
    bhalf* qbuf   = (bhalf*)carve((size_t)BH * NTOK * HD * 2);     // [64][3136][64]
    bhalf* kT     = (bhalf*)carve((size_t)BH * HD * NTOK * 2);     // [64][64][3136]
    bhalf* vT     = (bhalf*)carve((size_t)BH * HD * NTOK * 2);     // [64][64][3136]
    bhalf* kvT    = (bhalf*)carve((size_t)BH * HD * HD * 2);       // [64][64][64]
    float* ksum   = (float*)carve((size_t)BH * HD * 4);
    float* zbuf   = (float*)carve((size_t)BH * NTOK * 4);
    bhalf* attn   = (bhalf*)carve((size_t)MROWS * CDIM * 2);       // merged [b][n][c]

    // --- stage 0: precision conversion / weight transposes -----------------
    {
        int nElem = MROWS * CDIM;  // 12,845,056
        cvt_f2b_kernel<<<nElem / 256, 256, 0, stream>>>(x, xb, nElem);
        transpose_cvt_kernel<<<(CDIM * CDIM) / 256, 256, 0, stream>>>(Wq, wqkvT, CDIM, CDIM);
        transpose_cvt_kernel<<<(CDIM * 2 * CDIM) / 256, 256, 0, stream>>>(
            Wkv, wqkvT + (size_t)CDIM * CDIM, CDIM, 2 * CDIM);
        transpose_cvt_kernel<<<(CDIM * CDIM) / 256, 256, 0, stream>>>(Wproj, wprojT, CDIM, CDIM);
    }

    // --- stage 1: fused QKV GEMM + relu + head split/transposes ------------
    // 24 wave-cols (1536/64), 784 macro-rows (25088/32), 8 waves/block
    qkv_gemm_kernel<<<dim3(3, MROWS / 32), 256, 0, stream>>>(
        xb, wqkvT, pos_enc, qbuf, kT, vT);

    // --- stage 2: per-head reductions & small GEMMs -------------------------
    ksum_kernel<<<BH, 256, 0, stream>>>(kT, ksum);
    // 64 heads * 4 (32x32) macro-tiles = 256 waves = 32 blocks
    kvmat_kernel<<<32, 256, 0, stream>>>(kT, vT, kvT);
    z_kernel<<<BH, 256, 0, stream>>>(qbuf, ksum, zbuf);

    // --- stage 3: depthwise conv (writes attn), then out-GEMM accumulates ---
    dwconv_kernel<<<BH * HD, 256, 0, stream>>>(vT, dwc_w, dwc_b, attn);
    // 64 heads * 196 (32x32) macro-tiles = 12544 waves = 1568 blocks
    attnout_kernel<<<(BH * 196) / 8, 256, 0, stream>>>(qbuf, kvT, zbuf, attn);

    // --- stage 4: output projection -----------------------------------------
    proj_gemm_kernel<<<MROWS / 32, 256, 0, stream>>>(attn, wprojT, bproj, out);
}
